// CrossModalDecoderLayer_35012573397736
// MI455X (gfx1250) — compile-verified, matrix operand order checked by reading, not run on hardware
//
#include <hip/hip_runtime.h>
#include <hip/hip_bf16.h>
#include <math.h>

typedef __bf16 bf16;
typedef bf16 bf16x16 __attribute__((ext_vector_type(16)));
typedef bf16 bf16x8  __attribute__((ext_vector_type(8)));
typedef float f32x8  __attribute__((ext_vector_type(8)));

#define DEV __device__ __forceinline__

constexpr int cB = 4, cNT = 1024, cNI = 576, cDIM = 1536, cCDIM = 1024;
constexpr int cH = 12, cHK = 4, cHD = 128, cE = 4, cINTER = 6144;
constexpr int TOK = cB * cNT;   // 4096
constexpr int CTX = cB * cNI;   // 2304
constexpr float cEPS = 1e-6f;

// ---------------------------------------------------------------------------
// WMMA helpers (CDNA5 wave32, v_wmma_f32_16x16x32_bf16)
// ---------------------------------------------------------------------------
DEV f32x8 wmma_bf16(bf16x16 a, bf16x16 b, f32x8 c) {
  return __builtin_amdgcn_wmma_f32_16x16x32_bf16(false, a, false, b, (short)0,
                                                 c, false, false);
}

// Load a 16x32 bf16 A/B fragment. Per ISA layout: lanes 0-15 -> row = lane,
// K = {0..7,16..23}; lanes 16-31 -> row = lane-16, K = {8..15,24..31}.
DEV bf16x16 frag_ld(const bf16* p0, int stride, int lane) {
  int half = lane >> 4, r = lane & 15;
  const bf16* p = p0 + (size_t)r * stride + half * 8;
  bf16x8 lo = *(const bf16x8*)(p);
  bf16x8 hi = *(const bf16x8*)(p + 16);
  bf16x16 f;
#pragma unroll
  for (int i = 0; i < 8; ++i) { f[i] = lo[i]; f[i + 8] = hi[i]; }
  return f;
}

// ---------------------------------------------------------------------------
// GEMM tile config: 128x128x32 block tile, 8 waves (2x4), 64x32 per wave.
// LDS rows padded 32 -> 40 bf16 (80B, 16B aligned) for conflict-free b128.
// Software pipeline: global loads for tile kt+1 issued before the WMMAs of
// tile kt; loadcnt waits land in the convert+ds_store after compute.
// Double-buffered LDS -> one barrier per K-step.
// ---------------------------------------------------------------------------
constexpr int BM = 128, BN = 128, BK = 32, SSTR = 40;
constexpr int STILE = BM * SSTR;   // elements per LDS tile buffer

struct R4 { float4 v[4]; };
struct RB2 { bf16x8 v[2]; };

// ---- fp32 A tile (converted to bf16 on LDS store) ----
DEV R4 gload_A(const float* A, int lda, int row0, int k0) {
  R4 g;
  int tid = threadIdx.x;
#pragma unroll
  for (int i = 0; i < 4; ++i) {
    int f = tid + i * 256;               // 0..1023 float4 chunks
    int r = f >> 3, c4 = (f & 7) << 2;   // 8 float4 per 32-col row
    g.v[i] = *(const float4*)(A + (size_t)(row0 + r) * lda + k0 + c4);
  }
  return g;
}

DEV void store_A(const R4& g, bf16* sA) {
  int tid = threadIdx.x;
#pragma unroll
  for (int i = 0; i < 4; ++i) {
    int f = tid + i * 256;
    int r = f >> 3, c4 = (f & 7) << 2;
    bf16* d = sA + r * SSTR + c4;
    d[0] = (bf16)g.v[i].x; d[1] = (bf16)g.v[i].y;
    d[2] = (bf16)g.v[i].z; d[3] = (bf16)g.v[i].w;
  }
}

// ---- bf16 A tile (pure 16B copy, no conversion) ----
DEV RB2 gload_A16(const bf16* A, int lda, int row0, int k0) {
  RB2 g;
  int tid = threadIdx.x;
#pragma unroll
  for (int i = 0; i < 2; ++i) {
    int f = tid + i * 256;               // 0..511 chunks of 8 bf16
    int r = f >> 2, c8 = (f & 3) << 3;   // 4 chunks per 32-col row
    g.v[i] = *(const bf16x8*)(A + (size_t)(row0 + r) * lda + k0 + c8);
  }
  return g;
}

DEV void store_A16(const RB2& g, bf16* sA) {
  int tid = threadIdx.x;
#pragma unroll
  for (int i = 0; i < 2; ++i) {
    int f = tid + i * 256;
    int r = f >> 2, c8 = (f & 3) << 3;
    *(bf16x8*)(sA + r * SSTR + c8) = g.v[i];
  }
}

// ---- fp32 B tile (stored transposed in LDS: sB[n][k]) ----
DEV R4 gload_B(const float* B, int ldb, int k0, int col0) {
  R4 g;
  int tid = threadIdx.x;
#pragma unroll
  for (int i = 0; i < 4; ++i) {
    int f = tid + i * 256;
    int kr = f >> 5, c4 = (f & 31) << 2; // 32 float4 per 128-col row
    g.v[i] = *(const float4*)(B + (size_t)(k0 + kr) * ldb + col0 + c4);
  }
  return g;
}

DEV void store_B(const R4& g, bf16* sB) {
  int tid = threadIdx.x;
#pragma unroll
  for (int i = 0; i < 4; ++i) {
    int f = tid + i * 256;
    int kr = f >> 5, c4 = (f & 31) << 2;
    sB[(c4 + 0) * SSTR + kr] = (bf16)g.v[i].x;
    sB[(c4 + 1) * SSTR + kr] = (bf16)g.v[i].y;
    sB[(c4 + 2) * SSTR + kr] = (bf16)g.v[i].z;
    sB[(c4 + 3) * SSTR + kr] = (bf16)g.v[i].w;
  }
}

// Shared 64x32-per-wave WMMA compute step on current LDS buffers.
DEV void mma_step(const bf16* sA, const bf16* sB, int wm, int wn, int lane,
                  f32x8 acc[4][2]) {
  bf16x16 af[4], bfm[2];
#pragma unroll
  for (int m = 0; m < 4; ++m) af[m] = frag_ld(sA + (wm + m * 16) * SSTR, SSTR, lane);
#pragma unroll
  for (int n = 0; n < 2; ++n) bfm[n] = frag_ld(sB + (wn + n * 16) * SSTR, SSTR, lane);
#pragma unroll
  for (int m = 0; m < 4; ++m)
#pragma unroll
    for (int n = 0; n < 2; ++n) acc[m][n] = wmma_bf16(af[m], bfm[n], acc[m][n]);
}

// ---------------------------------------------------------------------------
// C = A @ B + bias   (fp32 out)
// ---------------------------------------------------------------------------
__global__ __launch_bounds__(256) void gemm_bias_kernel(
    const float* __restrict__ A, const float* __restrict__ Bw,
    const float* __restrict__ bias, float* __restrict__ C, int N, int K) {
  __shared__ bf16 sA[2][STILE];
  __shared__ bf16 sB[2][STILE];
  int row0 = blockIdx.y * BM, col0 = blockIdx.x * BN;
  int lane = threadIdx.x & 31, wave = threadIdx.x >> 5;
  int wm = (wave & 1) * 64, wn = (wave >> 1) * 32;
  f32x8 acc[4][2];
#pragma unroll
  for (int m = 0; m < 4; ++m)
#pragma unroll
    for (int n = 0; n < 2; ++n)
#pragma unroll
      for (int i = 0; i < 8; ++i) acc[m][n][i] = 0.f;
  int nk = K / BK;
  R4 ga = gload_A(A, K, row0, 0);
  R4 gb = gload_B(Bw, N, 0, col0);
  store_A(ga, sA[0]);
  store_B(gb, sB[0]);
  for (int kt = 0; kt < nk; ++kt) {
    __syncthreads();
    int cur = kt & 1;
    if (kt + 1 < nk) {
      ga = gload_A(A, K, row0, (kt + 1) * BK);
      gb = gload_B(Bw, N, (kt + 1) * BK, col0);
    }
    mma_step(sA[cur], sB[cur], wm, wn, lane, acc);
    if (kt + 1 < nk) {
      store_A(ga, sA[cur ^ 1]);
      store_B(gb, sB[cur ^ 1]);
    }
  }
  int half = lane >> 4, c = lane & 15;
#pragma unroll
  for (int m = 0; m < 4; ++m)
#pragma unroll
    for (int n = 0; n < 2; ++n) {
      int col = col0 + wn + n * 16 + c;
      int rb = row0 + wm + m * 16 + half * 8;
      float bb = bias[col];
#pragma unroll
      for (int r = 0; r < 8; ++r)
        C[(size_t)(rb + r) * N + col] = acc[m][n][r] + bb;
    }
}

// ---------------------------------------------------------------------------
// C = A @ B + bias   (bf16 out; used for the V projection)
// ---------------------------------------------------------------------------
__global__ __launch_bounds__(256) void gemm_biasb_kernel(
    const float* __restrict__ A, const float* __restrict__ Bw,
    const float* __restrict__ bias, bf16* __restrict__ C, int N, int K) {
  __shared__ bf16 sA[2][STILE];
  __shared__ bf16 sB[2][STILE];
  int row0 = blockIdx.y * BM, col0 = blockIdx.x * BN;
  int lane = threadIdx.x & 31, wave = threadIdx.x >> 5;
  int wm = (wave & 1) * 64, wn = (wave >> 1) * 32;
  f32x8 acc[4][2];
#pragma unroll
  for (int m = 0; m < 4; ++m)
#pragma unroll
    for (int n = 0; n < 2; ++n)
#pragma unroll
      for (int i = 0; i < 8; ++i) acc[m][n][i] = 0.f;
  int nk = K / BK;
  R4 ga = gload_A(A, K, row0, 0);
  R4 gb = gload_B(Bw, N, 0, col0);
  store_A(ga, sA[0]);
  store_B(gb, sB[0]);
  for (int kt = 0; kt < nk; ++kt) {
    __syncthreads();
    int cur = kt & 1;
    if (kt + 1 < nk) {
      ga = gload_A(A, K, row0, (kt + 1) * BK);
      gb = gload_B(Bw, N, (kt + 1) * BK, col0);
    }
    mma_step(sA[cur], sB[cur], wm, wn, lane, acc);
    if (kt + 1 < nk) {
      store_A(ga, sA[cur ^ 1]);
      store_B(gb, sB[cur ^ 1]);
    }
  }
  int half = lane >> 4, c = lane & 15;
#pragma unroll
  for (int m = 0; m < 4; ++m)
#pragma unroll
    for (int n = 0; n < 2; ++n) {
      int col = col0 + wn + n * 16 + c;
      int rb = row0 + wm + m * 16 + half * 8;
      float bb = bias[col];
#pragma unroll
      for (int r = 0; r < 8; ++r)
        C[(size_t)(rb + r) * N + col] = (bf16)(acc[m][n][r] + bb);
    }
}

// ---------------------------------------------------------------------------
// H = resid + gamma * (A @ B + bias)   (A is bf16 attention output)
// ---------------------------------------------------------------------------
__global__ __launch_bounds__(256) void gemm_oproj_kernel(
    const bf16* __restrict__ A, const float* __restrict__ Bw,
    const float* __restrict__ bias, const float* __restrict__ resid,
    const float* __restrict__ gamma, float* __restrict__ Hout, int N, int K) {
  __shared__ bf16 sA[2][STILE];
  __shared__ bf16 sB[2][STILE];
  int row0 = blockIdx.y * BM, col0 = blockIdx.x * BN;
  int lane = threadIdx.x & 31, wave = threadIdx.x >> 5;
  int wm = (wave & 1) * 64, wn = (wave >> 1) * 32;
  f32x8 acc[4][2];
#pragma unroll
  for (int m = 0; m < 4; ++m)
#pragma unroll
    for (int n = 0; n < 2; ++n)
#pragma unroll
      for (int i = 0; i < 8; ++i) acc[m][n][i] = 0.f;
  int nk = K / BK;
  RB2 ga = gload_A16(A, K, row0, 0);
  R4 gb = gload_B(Bw, N, 0, col0);
  store_A16(ga, sA[0]);
  store_B(gb, sB[0]);
  for (int kt = 0; kt < nk; ++kt) {
    __syncthreads();
    int cur = kt & 1;
    if (kt + 1 < nk) {
      ga = gload_A16(A, K, row0, (kt + 1) * BK);
      gb = gload_B(Bw, N, (kt + 1) * BK, col0);
    }
    mma_step(sA[cur], sB[cur], wm, wn, lane, acc);
    if (kt + 1 < nk) {
      store_A16(ga, sA[cur ^ 1]);
      store_B(gb, sB[cur ^ 1]);
    }
  }
  int half = lane >> 4, c = lane & 15;
#pragma unroll
  for (int m = 0; m < 4; ++m)
#pragma unroll
    for (int n = 0; n < 2; ++n) {
      int col = col0 + wn + n * 16 + c;
      int rb = row0 + wm + m * 16 + half * 8;
      float bb = bias[col], g = gamma[col];
#pragma unroll
      for (int r = 0; r < 8; ++r) {
        size_t idx = (size_t)(rb + r) * N + col;
        Hout[idx] = resid[idx] + g * (acc[m][n][r] + bb);
      }
    }
}

// ---------------------------------------------------------------------------
// act = silu(A @ Bg) * (A @ Bu)  -> bf16   (A is bf16; pure-copy staging)
// ---------------------------------------------------------------------------
__global__ __launch_bounds__(256) void gemm_gu_kernel(
    const bf16* __restrict__ A, const float* __restrict__ Bg,
    const float* __restrict__ Bu, bf16* __restrict__ act, int N, int K) {
  __shared__ bf16 sA[2][STILE];
  __shared__ bf16 sBg[2][STILE];
  __shared__ bf16 sBu[2][STILE];
  int row0 = blockIdx.y * BM, col0 = blockIdx.x * BN;
  int lane = threadIdx.x & 31, wave = threadIdx.x >> 5;
  int wm = (wave & 1) * 64, wn = (wave >> 1) * 32;
  f32x8 accg[4][2], accu[4][2];
#pragma unroll
  for (int m = 0; m < 4; ++m)
#pragma unroll
    for (int n = 0; n < 2; ++n)
#pragma unroll
      for (int i = 0; i < 8; ++i) { accg[m][n][i] = 0.f; accu[m][n][i] = 0.f; }
  int nk = K / BK;
  RB2 ga = gload_A16(A, K, row0, 0);
  R4 gg = gload_B(Bg, N, 0, col0);
  R4 gu = gload_B(Bu, N, 0, col0);
  store_A16(ga, sA[0]);
  store_B(gg, sBg[0]);
  store_B(gu, sBu[0]);
  for (int kt = 0; kt < nk; ++kt) {
    __syncthreads();
    int cur = kt & 1;
    if (kt + 1 < nk) {
      ga = gload_A16(A, K, row0, (kt + 1) * BK);
      gg = gload_B(Bg, N, (kt + 1) * BK, col0);
      gu = gload_B(Bu, N, (kt + 1) * BK, col0);
    }
    {
      bf16x16 af[4];
#pragma unroll
      for (int m = 0; m < 4; ++m)
        af[m] = frag_ld(sA[cur] + (wm + m * 16) * SSTR, SSTR, lane);
#pragma unroll
      for (int n = 0; n < 2; ++n) {
        bf16x16 bg = frag_ld(sBg[cur] + (wn + n * 16) * SSTR, SSTR, lane);
        bf16x16 bu = frag_ld(sBu[cur] + (wn + n * 16) * SSTR, SSTR, lane);
#pragma unroll
        for (int m = 0; m < 4; ++m) {
          accg[m][n] = wmma_bf16(af[m], bg, accg[m][n]);
          accu[m][n] = wmma_bf16(af[m], bu, accu[m][n]);
        }
      }
    }
    if (kt + 1 < nk) {
      store_A16(ga, sA[cur ^ 1]);
      store_B(gg, sBg[cur ^ 1]);
      store_B(gu, sBu[cur ^ 1]);
    }
  }
  int half = lane >> 4, c = lane & 15;
#pragma unroll
  for (int m = 0; m < 4; ++m)
#pragma unroll
    for (int n = 0; n < 2; ++n) {
      int col = col0 + wn + n * 16 + c;
      int rb = row0 + wm + m * 16 + half * 8;
#pragma unroll
      for (int r = 0; r < 8; ++r) {
        float g = accg[m][n][r];
        float s = g / (1.f + expf(-g));
        act[(size_t)(rb + r) * N + col] = (bf16)(s * accu[m][n][r]);
      }
    }
}

// ---------------------------------------------------------------------------
// ffn (+)= route[row][e] * (A @ B)   (A is bf16 act; routed combine)
// ---------------------------------------------------------------------------
__global__ __launch_bounds__(256) void gemm_wd_kernel(
    const bf16* __restrict__ A, const float* __restrict__ Bw,
    const float* __restrict__ route, int e, int first,
    float* __restrict__ ffn, int N, int K) {
  __shared__ bf16 sA[2][STILE];
  __shared__ bf16 sB[2][STILE];
  int row0 = blockIdx.y * BM, col0 = blockIdx.x * BN;
  int lane = threadIdx.x & 31, wave = threadIdx.x >> 5;
  int wm = (wave & 1) * 64, wn = (wave >> 1) * 32;
  f32x8 acc[4][2];
#pragma unroll
  for (int m = 0; m < 4; ++m)
#pragma unroll
    for (int n = 0; n < 2; ++n)
#pragma unroll
      for (int i = 0; i < 8; ++i) acc[m][n][i] = 0.f;
  int nk = K / BK;
  RB2 ga = gload_A16(A, K, row0, 0);
  R4 gb = gload_B(Bw, N, 0, col0);
  store_A16(ga, sA[0]);
  store_B(gb, sB[0]);
  for (int kt = 0; kt < nk; ++kt) {
    __syncthreads();
    int cur = kt & 1;
    if (kt + 1 < nk) {
      ga = gload_A16(A, K, row0, (kt + 1) * BK);
      gb = gload_B(Bw, N, (kt + 1) * BK, col0);
    }
    mma_step(sA[cur], sB[cur], wm, wn, lane, acc);
    if (kt + 1 < nk) {
      store_A16(ga, sA[cur ^ 1]);
      store_B(gb, sB[cur ^ 1]);
    }
  }
  int half = lane >> 4, c = lane & 15;
#pragma unroll
  for (int m = 0; m < 4; ++m)
#pragma unroll
    for (int n = 0; n < 2; ++n) {
      int col = col0 + wn + n * 16 + c;
      int rb = row0 + wm + m * 16 + half * 8;
#pragma unroll
      for (int r = 0; r < 8; ++r) {
        float w = route[(size_t)(rb + r) * cE + e];
        size_t idx = (size_t)(rb + r) * N + col;
        float v = w * acc[m][n][r];
        if (first) ffn[idx] = v; else ffn[idx] += v;
      }
    }
}

// ---------------------------------------------------------------------------
// Flash attention: 8 waves x 16 q-rows per block, 64-key blocks, WMMA both
// GEMMs; q/k/v all bf16; output bf16 (feeds o-projection A operand).
// ---------------------------------------------------------------------------
__global__ __launch_bounds__(256) void attn_kernel(
    const bf16* __restrict__ qn, const bf16* __restrict__ kn,
    const bf16* __restrict__ vb, const unsigned char* __restrict__ mask,
    bf16* __restrict__ outp) {
  __shared__ bf16 sV[128 * 72];       // V^T [d][key], stride 72 (144B, 16B-mult)
  __shared__ bf16 sP[8 * 16 * 72];    // per-wave P tile [qrow][key]
  int bh = blockIdx.y;
  int b = bh / cH, h = bh % cH, hk = h / (cH / cHK);
  int lane = threadIdx.x & 31, wave = threadIdx.x >> 5;
  int half = lane >> 4, c = lane & 15;
  int q0 = blockIdx.x * 128 + wave * 16;
  const bf16* qbase = qn + (size_t)(b * cNT + q0) * cDIM + h * cHD;
  bf16x16 aq[4];
#pragma unroll
  for (int d = 0; d < 4; ++d) aq[d] = frag_ld(qbase + d * 32, cDIM, lane);
  float m_i[8], l_i[8];
  f32x8 oacc[8];
#pragma unroll
  for (int r = 0; r < 8; ++r) { m_i[r] = -3.0e38f; l_i[r] = 0.f; }
#pragma unroll
  for (int d = 0; d < 8; ++d)
#pragma unroll
    for (int r = 0; r < 8; ++r) oacc[d][r] = 0.f;
  const float scale = 0.08838834764831845f;  // 1/sqrt(128)
  bf16* sPw = sP + wave * 16 * 72;

  for (int kb = 0; kb < cNI / 64; ++kb) {
    // stage V(64x128) bf16 [key][d] -> sV[d][key] (shared by all waves)
    bf16x8 vreg[4];
#pragma unroll
    for (int i = 0; i < 4; ++i) {
      int f = threadIdx.x + i * 256;   // 0..1023 chunks of 8 bf16
      int key = f >> 4, d8 = (f & 15) << 3;
      vreg[i] = *(const bf16x8*)(
          vb + (size_t)(b * cNI + kb * 64 + key) * (cHK * cHD) + hk * cHD + d8);
    }
#pragma unroll
    for (int i = 0; i < 4; ++i) {
      int f = threadIdx.x + i * 256;
      int key = f >> 4, d8 = (f & 15) << 3;
#pragma unroll
      for (int j = 0; j < 8; ++j) sV[(d8 + j) * 72 + key] = vreg[i][j];
    }
    __syncthreads();
    // S = Q K^T  (16 x 64), K fragments straight from global bf16
    f32x8 s[4];
#pragma unroll
    for (int n = 0; n < 4; ++n)
#pragma unroll
      for (int r = 0; r < 8; ++r) s[n][r] = 0.f;
#pragma unroll
    for (int d = 0; d < 4; ++d)
#pragma unroll
      for (int n = 0; n < 4; ++n) {
        const bf16* kb0 = kn + (size_t)(b * cNI + kb * 64 + n * 16) * (cHK * cHD)
                          + hk * cHD + d * 32;
        bf16x16 bk_ = frag_ld(kb0, cHK * cHD, lane);
        s[n] = wmma_bf16(aq[d], bk_, s[n]);
      }
    // scale + mask (mask depends only on key column)
#pragma unroll
    for (int n = 0; n < 4; ++n) {
      int key = kb * 64 + n * 16 + c;
      bool ok = mask[(size_t)b * cNI + key] != 0;
#pragma unroll
      for (int r = 0; r < 8; ++r) {
        float sv = s[n][r] * scale;
        s[n][r] = ok ? sv : -3.0e38f;
      }
    }
    // online softmax (rows striped across the 16-lane half-group)
    float mn[8], rs[8];
#pragma unroll
    for (int r = 0; r < 8; ++r) {
      float mx = fmaxf(fmaxf(s[0][r], s[1][r]), fmaxf(s[2][r], s[3][r]));
#pragma unroll
      for (int off = 8; off >= 1; off >>= 1) mx = fmaxf(mx, __shfl_xor(mx, off));
      mn[r] = fmaxf(m_i[r], mx);
    }
#pragma unroll
    for (int r = 0; r < 8; ++r) {
      float sum = 0.f;
#pragma unroll
      for (int n = 0; n < 4; ++n) {
        float p = expf(s[n][r] - mn[r]);
        s[n][r] = p;
        sum += p;
      }
#pragma unroll
      for (int off = 8; off >= 1; off >>= 1) sum += __shfl_xor(sum, off);
      rs[r] = sum;
    }
#pragma unroll
    for (int r = 0; r < 8; ++r) {
      float alpha = expf(m_i[r] - mn[r]);
      l_i[r] = l_i[r] * alpha + rs[r];
      m_i[r] = mn[r];
#pragma unroll
      for (int d = 0; d < 8; ++d) oacc[d][r] *= alpha;
    }
    // P (C-layout) -> LDS -> A-layout fragments
#pragma unroll
    for (int n = 0; n < 4; ++n)
#pragma unroll
      for (int r = 0; r < 8; ++r)
        sPw[(half * 8 + r) * 72 + n * 16 + c] = (bf16)s[n][r];
    __syncthreads();
    // O += P @ V
#pragma unroll
    for (int t = 0; t < 2; ++t) {
      bf16x16 ap = frag_ld(sPw + t * 32, 72, lane);
#pragma unroll
      for (int d = 0; d < 8; ++d) {
        bf16x16 bv = frag_ld(sV + (d * 16) * 72 + t * 32, 72, lane);
        oacc[d] = wmma_bf16(ap, bv, oacc[d]);
      }
    }
    __syncthreads();
  }
  bf16* ob = outp + (size_t)(b * cNT + q0) * cDIM + h * cHD;
#pragma unroll
  for (int d = 0; d < 8; ++d)
#pragma unroll
    for (int r = 0; r < 8; ++r)
      ob[(size_t)(half * 8 + r) * cDIM + d * 16 + c] =
          (bf16)(oacc[d][r] * (1.f / l_i[r]));
}

// ---------------------------------------------------------------------------
// Per-head (128-dim) RMSNorm -> bf16 (one wave per row)
// ---------------------------------------------------------------------------
__global__ __launch_bounds__(256) void headnorm_kernel(
    const float* __restrict__ x, const float* __restrict__ w,
    bf16* __restrict__ out, int rows) {
  int lane = threadIdx.x & 31, wave = threadIdx.x >> 5;
  int row = blockIdx.x * 8 + wave;
  if (row >= rows) return;
  const float4 v = *(const float4*)(x + (size_t)row * cHD + lane * 4);
  float ss = v.x * v.x + v.y * v.y + v.z * v.z + v.w * v.w;
#pragma unroll
  for (int off = 16; off >= 1; off >>= 1) ss += __shfl_xor(ss, off);
  float rms = rsqrtf(ss * (1.f / cHD) + cEPS);
  const float4 wv = *(const float4*)(w + lane * 4);
  bf16* o = out + (size_t)row * cHD + lane * 4;
  o[0] = (bf16)(v.x * rms * wv.x);
  o[1] = (bf16)(v.y * rms * wv.y);
  o[2] = (bf16)(v.z * rms * wv.z);
  o[3] = (bf16)(v.w * rms * wv.w);
}

// ---------------------------------------------------------------------------
// Row RMSNorm over DIM=1536 (one block per row)
// ---------------------------------------------------------------------------
__global__ __launch_bounds__(256) void rmsnorm_kernel(
    const float* __restrict__ x, const float* __restrict__ w,
    float* __restrict__ out) {
  int row = blockIdx.x, tid = threadIdx.x;
  int lane = tid & 31, wave = tid >> 5;
  const float* xr = x + (size_t)row * cDIM;
  float v[6];
  float ss = 0.f;
#pragma unroll
  for (int i = 0; i < 6; ++i) { v[i] = xr[tid + i * 256]; ss += v[i] * v[i]; }
#pragma unroll
  for (int off = 16; off >= 1; off >>= 1) ss += __shfl_xor(ss, off);
  __shared__ float sred[8];
  if (lane == 0) sred[wave] = ss;
  __syncthreads();
  float tot = 0.f;
#pragma unroll
  for (int i = 0; i < 8; ++i) tot += sred[i];
  float rms = rsqrtf(tot * (1.f / cDIM) + cEPS);
  float* orow = out + (size_t)row * cDIM;
#pragma unroll
  for (int i = 0; i < 6; ++i) {
    int d = tid + i * 256;
    orow[d] = v[i] * rms * w[d];
  }
}

// ---------------------------------------------------------------------------
// Row RMSNorm + gate logits + softmax top-2 routing (fused); t stored bf16
// ---------------------------------------------------------------------------
__global__ __launch_bounds__(256) void rmsnorm_gate_kernel(
    const float* __restrict__ x, const float* __restrict__ w,
    const float* __restrict__ wgate, bf16* __restrict__ tout,
    float* __restrict__ route) {
  int row = blockIdx.x, tid = threadIdx.x;
  int lane = tid & 31, wave = tid >> 5;
  const float* xr = x + (size_t)row * cDIM;
  float v[6];
  float ss = 0.f;
#pragma unroll
  for (int i = 0; i < 6; ++i) { v[i] = xr[tid + i * 256]; ss += v[i] * v[i]; }
#pragma unroll
  for (int off = 16; off >= 1; off >>= 1) ss += __shfl_xor(ss, off);
  __shared__ float sred[8];
  __shared__ float slog[4];
  if (lane == 0) sred[wave] = ss;
  if (tid < 4) slog[tid] = 0.f;
  __syncthreads();
  float tot = 0.f;
#pragma unroll
  for (int i = 0; i < 8; ++i) tot += sred[i];
  float rms = rsqrtf(tot * (1.f / cDIM) + cEPS);
  float lg[4] = {0.f, 0.f, 0.f, 0.f};
  bf16* orow = tout + (size_t)row * cDIM;
#pragma unroll
  for (int i = 0; i < 6; ++i) {
    int d = tid + i * 256;
    float t = v[i] * rms * w[d];
    orow[d] = (bf16)t;
#pragma unroll
    for (int e = 0; e < 4; ++e) lg[e] += t * wgate[d * 4 + e];
  }
#pragma unroll
  for (int e = 0; e < 4; ++e)
#pragma unroll
    for (int off = 16; off >= 1; off >>= 1) lg[e] += __shfl_xor(lg[e], off);
  if (lane == 0)
#pragma unroll
    for (int e = 0; e < 4; ++e) atomicAdd(&slog[e], lg[e]);
  __syncthreads();
  if (tid == 0) {
    float mx = fmaxf(fmaxf(slog[0], slog[1]), fmaxf(slog[2], slog[3]));
    float p[4], sum = 0.f;
#pragma unroll
    for (int e = 0; e < 4; ++e) { p[e] = expf(slog[e] - mx); sum += p[e]; }
#pragma unroll
    for (int e = 0; e < 4; ++e) p[e] /= sum;
    int i1 = 0;
#pragma unroll
    for (int e = 1; e < 4; ++e) if (p[e] > p[i1]) i1 = e;
    int i2 = (i1 == 0) ? 1 : 0;
#pragma unroll
    for (int e = 0; e < 4; ++e) if (e != i1 && p[e] > p[i2]) i2 = e;
    float sw = p[i1] + p[i2];
#pragma unroll
    for (int e = 0; e < 4; ++e)
      route[(size_t)row * 4 + e] = (e == i1) ? p[i1] / sw
                                 : (e == i2) ? p[i2] / sw : 0.f;
  }
}

// ---------------------------------------------------------------------------
// out = h + gamma_ffn * ffn
// ---------------------------------------------------------------------------
__global__ __launch_bounds__(256) void final_add_kernel(
    const float* __restrict__ h, const float* __restrict__ ffn,
    const float* __restrict__ gamma, float* __restrict__ out) {
  size_t i = (size_t)blockIdx.x * 256 + threadIdx.x;
  int col = (int)(i % cDIM);
  out[i] = h[i] + gamma[col] * ffn[i];
}

// ---------------------------------------------------------------------------
// Host launch
// ---------------------------------------------------------------------------
extern "C" void kernel_launch(void* const* d_in, const int* in_sizes, int n_in,
                              void* d_out, int out_size, void* d_ws,
                              size_t ws_size, hipStream_t stream) {
  (void)in_sizes; (void)n_in; (void)out_size; (void)ws_size;
  const float* hidden  = (const float*)d_in[0];
  const float* context = (const float*)d_in[1];
  const unsigned char* cmask = (const unsigned char*)d_in[2];
  const float* w_ln1 = (const float*)d_in[3];
  const float* w_ln2 = (const float*)d_in[4];
  const float* wq = (const float*)d_in[5];
  const float* bq = (const float*)d_in[6];
  const float* wk = (const float*)d_in[7];
  const float* bk = (const float*)d_in[8];
  const float* wv = (const float*)d_in[9];
  const float* bv = (const float*)d_in[10];
  const float* wo = (const float*)d_in[11];
  const float* bo = (const float*)d_in[12];
  const float* wqn = (const float*)d_in[13];
  const float* wkn = (const float*)d_in[14];
  const float* gca = (const float*)d_in[15];
  const float* gffn = (const float*)d_in[16];
  const float* wgate = (const float*)d_in[17];
  const float* w_g = (const float*)d_in[18];
  const float* w_u = (const float*)d_in[19];
  const float* w_d = (const float*)d_in[20];

  // workspace layout (phase-union overlap for the big activation buffers)
  constexpr size_t SZ_H   = (size_t)TOK * cDIM * 4;        // 25.2 MB
  constexpr size_t SZ_T   = (size_t)TOK * cDIM * 2;        // bf16 t
  constexpr size_t SZ_RT  = (size_t)TOK * cE * 4;
  constexpr size_t SZ_XN  = SZ_H;
  constexpr size_t SZ_Q   = SZ_H;
  constexpr size_t SZ_QN  = (size_t)TOK * cDIM * 2;
  constexpr size_t SZ_KR  = (size_t)CTX * (cHK * cHD) * 4;
  constexpr size_t SZ_KN  = (size_t)CTX * (cHK * cHD) * 2;
  constexpr size_t SZ_V   = (size_t)CTX * (cHK * cHD) * 2; // bf16 V
  constexpr size_t SZ_AO  = (size_t)TOK * cDIM * 2;        // bf16 attn out

  char* ws = (char*)d_ws;
  size_t oH = 0;
  size_t oT = oH + SZ_H;
  size_t oRT = oT + SZ_T;
  size_t oFFN = oRT + SZ_RT;
  size_t oU = oFFN + SZ_H;                 // phase union base
  size_t oXN = oU;
  size_t oQ  = oXN + SZ_XN;
  size_t oQN = oQ + SZ_Q;
  size_t oKR = oQN + SZ_QN;
  size_t oKN = oKR + SZ_KR;
  size_t oV  = oKN + SZ_KN;
  size_t oAO = oV + SZ_V;
  size_t oACT = oU;                        // overlaps dead phase-1 buffers
  (void)SZ_AO;

  float* hbuf  = (float*)(ws + oH);
  bf16*  tbuf  = (bf16*)(ws + oT);
  float* route = (float*)(ws + oRT);
  float* ffn   = (float*)(ws + oFFN);
  float* xn    = (float*)(ws + oXN);
  float* qbuf  = (float*)(ws + oQ);
  bf16*  qn    = (bf16*)(ws + oQN);
  float* kraw  = (float*)(ws + oKR);
  bf16*  knb   = (bf16*)(ws + oKN);
  bf16*  vbuf  = (bf16*)(ws + oV);
  bf16*  attno = (bf16*)(ws + oAO);
  bf16*  act   = (bf16*)(ws + oACT);
  float* outp  = (float*)d_out;

  // 1) ln1
  rmsnorm_kernel<<<TOK, 256, 0, stream>>>(hidden, w_ln1, xn);
  // 2) q/k/v projections (V emitted directly as bf16)
  gemm_bias_kernel<<<dim3(cDIM / BN, TOK / BM), 256, 0, stream>>>(
      xn, wq, bq, qbuf, cDIM, cDIM);
  gemm_bias_kernel<<<dim3((cHK * cHD) / BN, CTX / BM), 256, 0, stream>>>(
      context, wk, bk, kraw, cHK * cHD, cCDIM);
  gemm_biasb_kernel<<<dim3((cHK * cHD) / BN, CTX / BM), 256, 0, stream>>>(
      context, wv, bv, vbuf, cHK * cHD, cCDIM);
  // 3) q/k per-head RMSNorm -> bf16
  headnorm_kernel<<<(TOK * cH) / 8, 256, 0, stream>>>(qbuf, wqn, qn, TOK * cH);
  headnorm_kernel<<<(CTX * cHK) / 8, 256, 0, stream>>>(kraw, wkn, knb, CTX * cHK);
  // 4) flash attention (bf16 in, bf16 out)
  attn_kernel<<<dim3(cNT / 128, cB * cH), 256, 0, stream>>>(
      qn, knb, vbuf, cmask, attno);
  // 5) o-projection + layer-scaled residual (bf16 A)
  gemm_oproj_kernel<<<dim3(cDIM / BN, TOK / BM), 256, 0, stream>>>(
      attno, wo, bo, hidden, gca, hbuf, cDIM, cDIM);
  // 6) ln2 + gate + top-2 routing (t stored bf16)
  rmsnorm_gate_kernel<<<TOK, 256, 0, stream>>>(hbuf, w_ln2, wgate, tbuf, route);
  // 7) dense routed experts (bf16 activations end-to-end)
  for (int e = 0; e < cE; ++e) {
    const float* bg = w_g + (size_t)e * cDIM * cINTER;
    const float* bu = w_u + (size_t)e * cDIM * cINTER;
    const float* bd = w_d + (size_t)e * cINTER * cDIM;
    gemm_gu_kernel<<<dim3(cINTER / BN, TOK / BM), 256, 0, stream>>>(
        tbuf, bg, bu, act, cINTER, cDIM);
    gemm_wd_kernel<<<dim3(cDIM / BN, TOK / BM), 256, 0, stream>>>(
        act, bd, route, e, (e == 0) ? 1 : 0, ffn, cDIM, cINTER);
  }
  // 8) final residual
  final_add_kernel<<<(TOK * cDIM) / 256, 256, 0, stream>>>(hbuf, ffn, gffn, outp);
}
